// GDTLayer_5952824672823
// MI455X (gfx1250) — matrix operand order repaired and checked by reading, not compiled, not run on hardware
//
#include <hip/hip_runtime.h>

typedef __attribute__((ext_vector_type(16))) __bf16 v16bf;
typedef __attribute__((ext_vector_type(8)))  float  v8f;

#define DV 128
#define HH 8
#define DHV 16

// ---------------------------------------------------------------- utilities
__device__ __forceinline__ unsigned fenc(float f) {
    unsigned u = __float_as_uint(f);
    return (u & 0x80000000u) ? ~u : (u | 0x80000000u);
}
__device__ __forceinline__ float fdec(unsigned u) {
    unsigned v = (u & 0x80000000u) ? (u & 0x7FFFFFFFu) : ~u;
    return __uint_as_float(v);
}

// ------------------------------------------------------- f32 -> bf16 convert
__global__ __launch_bounds__(256) void k_cvt_bf16(const float* __restrict__ in,
                                                  __bf16* __restrict__ out, int n) {
    int i = blockIdx.x * 256 + threadIdx.x;
    if (i < n) out[i] = (__bf16)in[i];
}

// ------------------------------------------------------------- layer norm
// one wave32 per 128-wide row; optional f32 and bf16 outputs
__global__ __launch_bounds__(256) void k_ln(const float* __restrict__ x,
                                            const float* __restrict__ g,
                                            const float* __restrict__ b,
                                            float* __restrict__ outF,
                                            __bf16* __restrict__ outB, int rows) {
    int wave = threadIdx.x >> 5, lane = threadIdx.x & 31;
    int row  = blockIdx.x * 8 + wave;
    if (row >= rows) return;
    float4 v = ((const float4*)(x + (size_t)row * DV))[lane];
    float s  = v.x + v.y + v.z + v.w;
    float ss = v.x * v.x + v.y * v.y + v.z * v.z + v.w * v.w;
    #pragma unroll
    for (int i = 16; i > 0; i >>= 1) {
        s  += __shfl_xor(s, i, 32);
        ss += __shfl_xor(ss, i, 32);
    }
    float mu  = s * (1.0f / DV);
    float var = ss * (1.0f / DV) - mu * mu;
    float rs  = rsqrtf(var + 1e-5f);
    float4 gv = ((const float4*)g)[lane];
    float4 bv = ((const float4*)b)[lane];
    float4 o;
    o.x = (v.x - mu) * rs * gv.x + bv.x;
    o.y = (v.y - mu) * rs * gv.y + bv.y;
    o.z = (v.z - mu) * rs * gv.z + bv.z;
    o.w = (v.w - mu) * rs * gv.w + bv.w;
    if (outF) ((float4*)(outF + (size_t)row * DV))[lane] = o;
    if (outB) {
        union { __bf16 h[4]; ushort4 u; } t;
        t.h[0] = (__bf16)o.x; t.h[1] = (__bf16)o.y;
        t.h[2] = (__bf16)o.z; t.h[3] = (__bf16)o.w;
        *(ushort4*)(outB + (size_t)row * DV + lane * 4) = t.u;
    }
}

// -------------------------------------------------- WMMA bf16 GEMM: C = A * W^T
// A: M x K (row major bf16), W: Nout x K (row major bf16)
// epilogue: +bias[col], optional relu, optional +resid, store f32 and/or bf16
__global__ __launch_bounds__(256) void k_gemm_wmma(const __bf16* __restrict__ A,
                                                   const __bf16* __restrict__ W,
                                                   const float* __restrict__ bias,
                                                   const float* __restrict__ resid,
                                                   float* __restrict__ outF,
                                                   __bf16* __restrict__ outB,
                                                   int M, int K, int Nout, int relu) {
    int wave = threadIdx.x >> 5, lane = threadIdx.x & 31;
    int colTiles = Nout >> 4;
    int rowTiles = (M + 15) >> 4;
    int tile = blockIdx.x * 8 + wave;
    if (tile >= rowTiles * colTiles) return;   // uniform per wave
    int rowT = tile / colTiles, colT = tile - rowT * colTiles;
    int m0 = rowT << 4, n0 = colT << 4;

    int rowA = m0 + (lane & 15);
    int rowAc = rowA < M ? rowA : M - 1;       // clamp reads
    int aK = (lane >> 4) << 3;                 // 0 or 8
    int colW = n0 + (lane & 15);
    int bK = (lane >> 4) << 3;                 // 0 or 8

    v8f acc = {};
    for (int k0 = 0; k0 < K; k0 += 32) {
        union { v16bf v; uint4 q[2]; } ua, ub;
        const __bf16* ap = A + (size_t)rowAc * K + k0 + aK;
        ua.q[0] = *(const uint4*)ap;
        ua.q[1] = *(const uint4*)(ap + 16);
        const __bf16* bp = W + (size_t)colW * K + k0 + bK;
        ub.q[0] = *(const uint4*)bp;
        ub.q[1] = *(const uint4*)(bp + 16);
        acc = __builtin_amdgcn_wmma_f32_16x16x32_bf16(
            false, ua.v, false, ub.v, (short)0, acc, false, false);
    }

    int col   = n0 + (lane & 15);
    int rbase = m0 + ((lane >> 4) << 3);
    float bv = bias ? bias[col] : 0.0f;
    #pragma unroll
    for (int r = 0; r < 8; ++r) {
        int row = rbase + r;
        if (row >= M) break;
        float v = acc[r] + bv;
        if (relu) v = v > 0.0f ? v : 0.0f;
        size_t idx = (size_t)row * Nout + col;
        if (resid) v += resid[idx];
        if (outF) outF[idx] = v;
        if (outB) outB[idx] = (__bf16)v;
    }
}

// --------------------------------------------- per-node per-head attn logits
__global__ __launch_bounds__(256) void k_eh_et(const float* __restrict__ feat,
                                               const float* __restrict__ attn_h,
                                               const float* __restrict__ attn_t,
                                               float* __restrict__ eh,
                                               float* __restrict__ et, int N) {
    int idx = blockIdx.x * 256 + threadIdx.x;
    if (idx >= N * HH) return;
    int n = idx >> 3, h = idx & 7;
    const float4* fp = (const float4*)(feat + (size_t)n * DV + h * DHV);
    const float4* ah = (const float4*)(attn_h + h * DHV);
    const float4* at = (const float4*)(attn_t + h * DHV);
    float s1 = 0.0f, s2 = 0.0f;
    #pragma unroll
    for (int j = 0; j < 4; ++j) {
        float4 f = fp[j], a = ah[j], t = at[j];
        s1 += f.x * a.x + f.y * a.y + f.z * a.z + f.w * a.w;
        s2 += f.x * t.x + f.y * t.y + f.z * t.z + f.w * t.w;
    }
    eh[idx] = s1;
    et[idx] = s2;
}

__global__ __launch_bounds__(256) void k_init_softmax(unsigned* __restrict__ emax,
                                                      float* __restrict__ asum, int n) {
    int i = blockIdx.x * 256 + threadIdx.x;
    if (i < n) { emax[i] = 0x007FFFFFu; /* enc(-inf) */ asum[i] = 0.0f; }
}

// ----------------------------------- edge logits + leaky relu + segment max
__global__ __launch_bounds__(256) void k_edge_logits(const int* __restrict__ src,
                                                     const int* __restrict__ dst,
                                                     const float* __restrict__ eh,
                                                     const float* __restrict__ et,
                                                     float* __restrict__ a,
                                                     unsigned* __restrict__ emax, int E) {
    int e = blockIdx.x * 256 + threadIdx.x;
    if (e >= E) return;
    int s = src[e], d = dst[e];
    const float* ehs = eh + (size_t)s * HH;
    const float* etd = et + (size_t)d * HH;
    #pragma unroll
    for (int h = 0; h < HH; ++h) {
        float v = ehs[h] + etd[h];
        v = v > 0.0f ? v : 0.2f * v;
        a[(size_t)e * HH + h] = v;
        atomicMax(emax + (size_t)d * HH + h, fenc(v));
    }
}

// --------------------------------------------------- exp(e - max) + segment sum
__global__ __launch_bounds__(256) void k_edge_exp(const int* __restrict__ dst,
                                                  float* __restrict__ a,
                                                  const unsigned* __restrict__ emax,
                                                  float* __restrict__ asum, int E) {
    int idx = blockIdx.x * 256 + threadIdx.x;
    if (idx >= E * HH) return;
    int e = idx >> 3, h = idx & 7;
    int d = dst[e];
    float m = fdec(emax[(size_t)d * HH + h]);
    float v = expf(a[idx] - m);
    a[idx] = v;
    atomicAdd(asum + (size_t)d * HH + h, v);
}

__global__ __launch_bounds__(256) void k_edge_norm(const int* __restrict__ dst,
                                                   float* __restrict__ a,
                                                   const float* __restrict__ asum, int E) {
    int idx = blockIdx.x * 256 + threadIdx.x;
    if (idx >= E * HH) return;
    int e = idx >> 3, h = idx & 7;
    a[idx] = a[idx] / asum[(size_t)dst[e] * HH + h];
}

// ----------------------------------------- hop scatter: agg[dst] += feat[src]*a
// one wave32 per edge; lane handles 4 consecutive floats (head = lane>>2)
__global__ __launch_bounds__(256) void k_scatter(const int* __restrict__ src,
                                                 const int* __restrict__ dst,
                                                 const float* __restrict__ a,
                                                 const float* __restrict__ feat,
                                                 float* __restrict__ agg, int E) {
    int wave = threadIdx.x >> 5, lane = threadIdx.x & 31;
    int e = blockIdx.x * 8 + wave;
    if (e >= E) return;
    int s = src[e], d = dst[e];
    float av = a[(size_t)e * HH + (lane >> 2)];
    float4 f = ((const float4*)(feat + (size_t)s * DV))[lane];
    float* o = agg + (size_t)d * DV + lane * 4;
    atomicAdd(o + 0, f.x * av);
    atomicAdd(o + 1, f.y * av);
    atomicAdd(o + 2, f.z * av);
    atomicAdd(o + 3, f.w * av);
}

// feat = (1-alpha)*agg + alpha*feat0      (float4 granularity)
__global__ __launch_bounds__(256) void k_combine(const float* __restrict__ agg,
                                                 const float* __restrict__ feat0,
                                                 float* __restrict__ feat, int n4) {
    int i = blockIdx.x * 256 + threadIdx.x;
    if (i >= n4) return;
    float4 g = ((const float4*)agg)[i];
    float4 f = ((const float4*)feat0)[i];
    float4 o;
    o.x = 0.85f * g.x + 0.15f * f.x;
    o.y = 0.85f * g.y + 0.15f * f.y;
    o.z = 0.85f * g.z + 0.15f * f.z;
    o.w = 0.85f * g.w + 0.15f * f.w;
    ((float4*)feat)[i] = o;
}

// rst = feat + h
__global__ __launch_bounds__(256) void k_add(const float* __restrict__ x,
                                             const float* __restrict__ y,
                                             float* __restrict__ out, int n4) {
    int i = blockIdx.x * 256 + threadIdx.x;
    if (i >= n4) return;
    float4 a = ((const float4*)x)[i];
    float4 b = ((const float4*)y)[i];
    float4 o = {a.x + b.x, a.y + b.y, a.z + b.z, a.w + b.w};
    ((float4*)out)[i] = o;
}

// ===========================================================================
extern "C" void kernel_launch(void* const* d_in, const int* in_sizes, int n_in,
                              void* d_out, int out_size, void* d_ws, size_t ws_size,
                              hipStream_t stream) {
    const float* ent   = (const float*)d_in[0];
    const int*   src   = (const int*)d_in[1];
    const int*   dst   = (const int*)d_in[2];
    const float* W_ent = (const float*)d_in[3];
    const float* attnh = (const float*)d_in[4];
    const float* attnt = (const float*)d_in[5];
    const float* ln1g  = (const float*)d_in[6];
    const float* ln1b  = (const float*)d_in[7];
    const float* ln2g  = (const float*)d_in[8];
    const float* ln2b  = (const float*)d_in[9];
    const float* w1    = (const float*)d_in[10];
    const float* b1    = (const float*)d_in[11];
    const float* w2    = (const float*)d_in[12];
    const float* b2    = (const float*)d_in[13];

    const int N = in_sizes[0] / DV;
    const int E = in_sizes[1];
    const int D4 = 4 * DV;          // 512

    // ---- carve workspace
    char* base = (char*)d_ws;
    size_t off = 0;
    auto carve = [&](size_t bytes) -> void* {
        void* r = base + off;
        off += (bytes + 255) & ~(size_t)255;
        return r;
    };
    float*  h       = (float*)carve((size_t)N * DV * 4);
    __bf16* hbf     = (__bf16*)carve((size_t)N * DV * 2);   // reused as x_bf16
    float*  feat0   = (float*)carve((size_t)N * DV * 4);
    float*  featc   = (float*)carve((size_t)N * DV * 4);
    float*  agg     = (float*)carve((size_t)N * DV * 4);    // reused as rst
    float*  aw      = (float*)carve((size_t)E * HH * 4);
    float*  eh      = (float*)carve((size_t)N * HH * 4);
    float*  et      = (float*)carve((size_t)N * HH * 4);
    unsigned* emax  = (unsigned*)carve((size_t)N * HH * 4);
    float*  asum    = (float*)carve((size_t)N * HH * 4);
    __bf16* wentbf  = (__bf16*)carve((size_t)DV * DV * 2);
    __bf16* w1bf    = (__bf16*)carve((size_t)D4 * DV * 2);
    __bf16* w2bf    = (__bf16*)carve((size_t)DV * D4 * 2);
    __bf16* y1bf    = (__bf16*)carve((size_t)N * D4 * 2);
    float*  out     = (float*)d_out;

    // ---- weight conversions to bf16
    k_cvt_bf16<<<(DV * DV + 255) / 256, 256, 0, stream>>>(W_ent, wentbf, DV * DV);
    k_cvt_bf16<<<(D4 * DV + 255) / 256, 256, 0, stream>>>(w1, w1bf, D4 * DV);
    k_cvt_bf16<<<(DV * D4 + 255) / 256, 256, 0, stream>>>(w2, w2bf, DV * D4);

    // ---- LN1 -> h (f32) and h (bf16)
    k_ln<<<(N + 7) / 8, 256, 0, stream>>>(ent, ln1g, ln1b, h, hbf, N);

    // ---- feat0 = h @ W_ent^T   (WMMA bf16)
    {
        int tiles = ((N + 15) / 16) * (DV / 16);
        k_gemm_wmma<<<(tiles + 7) / 8, 256, 0, stream>>>(
            hbf, wentbf, nullptr, nullptr, feat0, nullptr, N, DV, DV, 0);
    }

    // ---- attention logits & edge softmax
    k_eh_et<<<(N * HH + 255) / 256, 256, 0, stream>>>(feat0, attnh, attnt, eh, et, N);
    k_init_softmax<<<(N * HH + 255) / 256, 256, 0, stream>>>(emax, asum, N * HH);
    k_edge_logits<<<(E + 255) / 256, 256, 0, stream>>>(src, dst, eh, et, aw, emax, E);
    k_edge_exp<<<(E * HH + 255) / 256, 256, 0, stream>>>(dst, aw, emax, asum, E);
    k_edge_norm<<<(E * HH + 255) / 256, 256, 0, stream>>>(dst, aw, asum, E);

    // ---- 6 propagation hops
    const int n4 = N * DV / 4;
    for (int hop = 0; hop < 6; ++hop) {
        hipMemsetAsync(agg, 0, (size_t)N * DV * 4, stream);
        const float* rd = (hop == 0) ? feat0 : featc;
        k_scatter<<<(E + 7) / 8, 256, 0, stream>>>(src, dst, aw, rd, agg, E);
        k_combine<<<(n4 + 255) / 256, 256, 0, stream>>>(agg, feat0, featc, n4);
    }

    // ---- rst = feat + h   (reuse agg as rst)
    float* rst = agg;
    k_add<<<(n4 + 255) / 256, 256, 0, stream>>>(featc, h, rst, n4);

    // ---- LN2 -> x (bf16, reuse hbf)
    k_ln<<<(N + 7) / 8, 256, 0, stream>>>(rst, ln2g, ln2b, nullptr, hbf, N);

    // ---- FFN1: y1 = relu(x @ w1^T + b1) -> bf16
    {
        int tiles = ((N + 15) / 16) * (D4 / 16);
        k_gemm_wmma<<<(tiles + 7) / 8, 256, 0, stream>>>(
            hbf, w1bf, b1, nullptr, nullptr, y1bf, N, DV, D4, 1);
    }
    // ---- FFN2 + residual: out = y1 @ w2^T + b2 + rst
    {
        int tiles = ((N + 15) / 16) * (DV / 16);
        k_gemm_wmma<<<(tiles + 7) / 8, 256, 0, stream>>>(
            y1bf, w2bf, b2, rst, out, nullptr, N, D4, DV, 0);
    }
}
